// AttentionPolicy_28329604285076
// MI455X (gfx1250) — matrix operand order repaired and checked by reading, not compiled, not run on hardware
//
#include <hip/hip_runtime.h>
#include <hip/hip_bf16.h>

// ---------------------------------------------------------------------------
// Sizes from the reference
// ---------------------------------------------------------------------------
#define AN 859
#define BN 64
#define DN 128
#define HN 256
#define ON 8
// N = AN*BN rows; attention over 5 tokens, 4 heads x dh=64, 1/SCALE = 0.125

typedef __attribute__((ext_vector_type(16))) __bf16 v16bf;
typedef __attribute__((ext_vector_type(8)))  float  v8f;
typedef __attribute__((ext_vector_type(4)))  int    v4i;

union FragAB {
    v16bf v;
    uint4 q[2];
};

// CDNA5 async global->LDS path (bypasses VGPRs, tracked by ASYNCcnt)
#if __has_builtin(__builtin_amdgcn_global_load_async_to_lds_b128)
#define HAVE_ASYNC_LDS 1
typedef __attribute__((address_space(1))) v4i gv4i;
typedef __attribute__((address_space(3))) v4i lv4i;
#endif

__device__ __forceinline__ void wait_async_lds()
{
#ifdef HAVE_ASYNC_LDS
#if __has_builtin(__builtin_amdgcn_s_wait_asynccnt)
    __builtin_amdgcn_s_wait_asynccnt(0);
#else
    asm volatile("s_wait_asynccnt 0" ::: "memory");
#endif
#endif
}

// ---------------------------------------------------------------------------
// Prep: f32 -> bf16 weight conversion
// ---------------------------------------------------------------------------
__global__ __launch_bounds__(256) void cvt_bf16_kernel(
    const float* __restrict__ src, __bf16* __restrict__ dst, int n)
{
    int i = blockIdx.x * 256 + threadIdx.x;
    if (i < n) dst[i] = (__bf16)src[i];
}

// ---------------------------------------------------------------------------
// Prep: three eval-mode batchnorms of obs, stored as bf16
// ---------------------------------------------------------------------------
__global__ __launch_bounds__(256) void bn_prep_kernel(
    const float* __restrict__ obs,
    const float* __restrict__ qm, const float* __restrict__ qv,
    const float* __restrict__ km, const float* __restrict__ kv,
    const float* __restrict__ vm, const float* __restrict__ vv,
    __bf16* __restrict__ bnq, __bf16* __restrict__ bnk, __bf16* __restrict__ bnv,
    int n)
{
    int i = blockIdx.x * 256 + threadIdx.x;
    if (i < n) {
        int d = i & (DN - 1);
        float x = obs[i];
        bnq[i] = (__bf16)((x - qm[d]) * rsqrtf(qv[d] + 1e-5f));
        bnk[i] = (__bf16)((x - km[d]) * rsqrtf(kv[d] + 1e-5f));
        bnv[i] = (__bf16)((x - vm[d]) * rsqrtf(vv[d] + 1e-5f));
    }
}

// ---------------------------------------------------------------------------
// Main fused kernel: one workgroup = (agent, 16-row batch tile)
//   8 waves; each wave owns 2 of the 16 N-tiles (columns) of H=256.
// ---------------------------------------------------------------------------
__global__ __launch_bounds__(256) void attn_main_kernel(
    const __bf16* __restrict__ bnq, const __bf16* __restrict__ bnk,
    const __bf16* __restrict__ bnv,
    const __bf16* __restrict__ wq, const __bf16* __restrict__ wk,
    const __bf16* __restrict__ wv, const __bf16* __restrict__ wfc,
    const int*    __restrict__ cloest,
    const float*  __restrict__ bq, const float* __restrict__ bk,
    const float*  __restrict__ bv, const float* __restrict__ bfc,
    const float*  __restrict__ Wl1, const float* __restrict__ bl1,
    float* __restrict__ out)
{
    const int blk  = blockIdx.x;
    const int a    = blk >> 2;          // agent
    const int b0   = (blk & 3) * 16;    // batch-tile base
    const int tid  = threadIdx.x;
    const int lane = tid & 31;
    const int wid  = tid >> 5;          // 0..7
    const int nt0  = wid * 2;           // first of 2 owned N-tiles

    // LDS (about 38 KB total)
    __shared__ __align__(16) __bf16 Xtile[16][136];   // input tile, padded rows
    __shared__ __align__(16) __bf16 Qs[16][264];      // Q, later reused as x
    __shared__ __align__(16) __bf16 Ktmp[16][264];    // K_t / V_t / fc out
    __shared__ float xacc[16][260];                   // f32 attention output acc
    __shared__ float attnw[16][4][5];                 // energies -> weights
    __shared__ int   nbidx[5];

    if (tid < 5) nbidx[tid] = cloest[a * 5 + tid];
    __syncthreads();

    // Stage a 16x128 bf16 tile (one neighbor) into LDS: 256 threads x 16B.
    // Uses CDNA5 async global->LDS DMA when available.
    auto load_tile = [&](const __bf16* src, int nb) {
        int row = tid >> 4, seg = tid & 15;
        const __bf16* p = src + ((size_t)(nb * BN + b0 + row)) * DN + seg * 8;
#ifdef HAVE_ASYNC_LDS
        __builtin_amdgcn_global_load_async_to_lds_b128(
            (gv4i*)p, (lv4i*)&Xtile[row][seg * 8], 0, 0);
#else
        *(uint4*)&Xtile[row][seg * 8] = *(const uint4*)p;
#endif
    };

    // 16x128 @ 128x256 projection via WMMA; dst gets bias added, cast to bf16
    auto proj128 = [&](const __bf16* W, const float* bias, __bf16 (*dst)[264]) {
        v8f acc0 = {}, acc1 = {};
        const int colL = lane & 15;
        const int kb   = (lane >> 4) * 8;    // B-frag K base
        const int ka   = (lane >> 4) * 8;    // A-frag K base
        const int rowA = lane & 15;
        for (int ks = 0; ks < 4; ++ks) {
            const int k0 = ks * 32;
            FragAB af;
            af.q[0] = *(const uint4*)&Xtile[rowA][k0 + ka];
            af.q[1] = *(const uint4*)&Xtile[rowA][k0 + ka + 16];
            const __bf16* pb0 = W + (size_t)(nt0 * 16 + colL) * DN + k0 + kb;
            FragAB bf0, bf1;
            bf0.q[0] = *(const uint4*)pb0;
            bf0.q[1] = *(const uint4*)(pb0 + 16);
            const __bf16* pb1 = pb0 + 16 * DN;
            bf1.q[0] = *(const uint4*)pb1;
            bf1.q[1] = *(const uint4*)(pb1 + 16);
            acc0 = __builtin_amdgcn_wmma_f32_16x16x32_bf16(
                false, af.v, false, bf0.v, (short)0, acc0, false, false);
            acc1 = __builtin_amdgcn_wmma_f32_16x16x32_bf16(
                false, af.v, false, bf1.v, (short)0, acc1, false, false);
        }
        // C layout: lane&15 = col, row = vgpr + 8*(lane>>4)
        const int rb = (lane >> 4) * 8;
        const int c0 = nt0 * 16 + colL, c1 = c0 + 16;
        const float bb0 = bias[c0], bb1 = bias[c1];
        for (int v = 0; v < 8; ++v) {
            dst[rb + v][c0] = (__bf16)(acc0[v] + bb0);
            dst[rb + v][c1] = (__bf16)(acc1[v] + bb1);
        }
    };

    // ---- Q projection (token 0 only — only q-row 0 survives the slice) ----
    load_tile(bnq, nbidx[0]);
    wait_async_lds();
    __syncthreads();
    proj128(wq, bq, Qs);
    __syncthreads();

    // ---- K tokens: energies[row][head][t] = (Q . K_t)/sqrt(dh) ----
    for (int t = 0; t < 5; ++t) {
        load_tile(bnk, nbidx[t]);
        wait_async_lds();
        __syncthreads();
        proj128(wk, bk, Ktmp);
        __syncthreads();
        if (tid < 64) {
            int row = tid & 15, h = tid >> 4;
            float s = 0.f;
            for (int j = 0; j < 64; ++j)
                s += (float)Qs[row][h * 64 + j] * (float)Ktmp[row][h * 64 + j];
            attnw[row][h][t] = s * 0.125f;   // 1/sqrt(64)
        }
        __syncthreads();
    }

    // ---- softmax over tokens + zero the x accumulator ----
    if (tid < 64) {
        int row = tid & 15, h = tid >> 4;
        float m = attnw[row][h][0];
        for (int t = 1; t < 5; ++t) m = fmaxf(m, attnw[row][h][t]);
        float e[5], ssum = 0.f;
        for (int t = 0; t < 5; ++t) { e[t] = __expf(attnw[row][h][t] - m); ssum += e[t]; }
        float inv = 1.f / ssum;
        for (int t = 0; t < 5; ++t) attnw[row][h][t] = e[t] * inv;
    }
    for (int i = tid; i < 16 * 256; i += 256) xacc[i >> 8][i & 255] = 0.f;
    __syncthreads();

    // ---- V tokens: x += attn * V_t ----
    for (int t = 0; t < 5; ++t) {
        load_tile(bnv, nbidx[t]);
        wait_async_lds();
        __syncthreads();
        proj128(wv, bv, Ktmp);               // V_t in Ktmp
        __syncthreads();
        {
            const int c = tid, h = c >> 6;
            for (int row = 0; row < 16; ++row)
                xacc[row][c] += attnw[row][h][t] * (float)Ktmp[row][c];
        }
        __syncthreads();
    }

    // ---- x -> bf16 (reuse Qs) ----
    for (int i = tid; i < 16 * 256; i += 256)
        Qs[i >> 8][i & 255] = (__bf16)xacc[i >> 8][i & 255];
    __syncthreads();

    // ---- fc: 16x256 @ 256x256, bias + leaky-relu, bf16 into Ktmp ----
    {
        v8f acc0 = {}, acc1 = {};
        const int colL = lane & 15;
        const int kb = (lane >> 4) * 8, ka = (lane >> 4) * 8, rowA = lane & 15;
        for (int ks = 0; ks < 8; ++ks) {
            const int k0 = ks * 32;
            FragAB af;
            af.q[0] = *(const uint4*)&Qs[rowA][k0 + ka];
            af.q[1] = *(const uint4*)&Qs[rowA][k0 + ka + 16];
            const __bf16* pb0 = wfc + (size_t)(nt0 * 16 + colL) * HN + k0 + kb;
            FragAB bf0, bf1;
            bf0.q[0] = *(const uint4*)pb0;
            bf0.q[1] = *(const uint4*)(pb0 + 16);
            const __bf16* pb1 = pb0 + 16 * HN;
            bf1.q[0] = *(const uint4*)pb1;
            bf1.q[1] = *(const uint4*)(pb1 + 16);
            acc0 = __builtin_amdgcn_wmma_f32_16x16x32_bf16(
                false, af.v, false, bf0.v, (short)0, acc0, false, false);
            acc1 = __builtin_amdgcn_wmma_f32_16x16x32_bf16(
                false, af.v, false, bf1.v, (short)0, acc1, false, false);
        }
        const int rb = (lane >> 4) * 8;
        const int c0 = nt0 * 16 + colL, c1 = c0 + 16;
        const float bb0 = bfc[c0], bb1 = bfc[c1];
        for (int v = 0; v < 8; ++v) {
            float x0 = acc0[v] + bb0; x0 = x0 > 0.f ? x0 : 0.01f * x0;
            float x1 = acc1[v] + bb1; x1 = x1 > 0.f ? x1 : 0.01f * x1;
            Ktmp[rb + v][c0] = (__bf16)x0;
            Ktmp[rb + v][c1] = (__bf16)x1;
        }
    }
    __syncthreads();

    // ---- l1: 256 -> 8 (tiny, VALU) + store ----
    if (tid < 128) {
        int row = tid >> 3, oc = tid & 7;
        float s = bl1[oc];
        const float* wrow = Wl1 + oc * HN;
        for (int j = 0; j < HN; ++j)
            s += (float)Ktmp[row][j] * wrow[j];
        size_t n = (size_t)a * BN + b0 + row;
        out[n * ON + oc] = s;
    }
}

// ---------------------------------------------------------------------------
// Launch
// ---------------------------------------------------------------------------
extern "C" void kernel_launch(void* const* d_in, const int* in_sizes, int n_in,
                              void* d_out, int out_size, void* d_ws, size_t ws_size,
                              hipStream_t stream)
{
    const float* obs    = (const float*)d_in[0];
    const int*   cloest = (const int*)  d_in[1];
    const float* q_mean = (const float*)d_in[2];
    const float* q_var  = (const float*)d_in[3];
    const float* k_mean = (const float*)d_in[4];
    const float* k_var  = (const float*)d_in[5];
    const float* v_mean = (const float*)d_in[6];
    const float* v_var  = (const float*)d_in[7];
    const float* Wq     = (const float*)d_in[8];
    const float* bq     = (const float*)d_in[9];
    const float* Wk     = (const float*)d_in[10];
    const float* bk     = (const float*)d_in[11];
    const float* Wv     = (const float*)d_in[12];
    const float* bv     = (const float*)d_in[13];
    const float* Wfc    = (const float*)d_in[14];
    const float* bfc    = (const float*)d_in[15];
    const float* Wl1    = (const float*)d_in[16];
    const float* bl1    = (const float*)d_in[17];
    float* out = (float*)d_out;

    const size_t nABD = (size_t)AN * BN * DN;   // 7,036,928
    __bf16* bnq = (__bf16*)d_ws;
    __bf16* bnk = bnq + nABD;
    __bf16* bnv = bnk + nABD;
    __bf16* wqb = bnv + nABD;
    __bf16* wkb = wqb + HN * DN;
    __bf16* wvb = wkb + HN * DN;
    __bf16* wfcb = wvb + HN * DN;

    cvt_bf16_kernel<<<(HN * DN + 255) / 256, 256, 0, stream>>>(Wq, wqb, HN * DN);
    cvt_bf16_kernel<<<(HN * DN + 255) / 256, 256, 0, stream>>>(Wk, wkb, HN * DN);
    cvt_bf16_kernel<<<(HN * DN + 255) / 256, 256, 0, stream>>>(Wv, wvb, HN * DN);
    cvt_bf16_kernel<<<(HN * HN + 255) / 256, 256, 0, stream>>>(Wfc, wfcb, HN * HN);

    bn_prep_kernel<<<(int)((nABD + 255) / 256), 256, 0, stream>>>(
        obs, q_mean, q_var, k_mean, k_var, v_mean, v_var,
        bnq, bnk, bnv, (int)nABD);

    attn_main_kernel<<<AN * 4, 256, 0, stream>>>(
        bnq, bnk, bnv, wqb, wkb, wvb, wfcb, cloest,
        bq, bk, bv, bfc, Wl1, bl1, out);
}
